// Bilinear_Att_66185446031425
// MI455X (gfx1250) — compile-verified, matrix operand order checked by reading, not run on hardware
//
#include <hip/hip_runtime.h>
#include <hip/hip_bf16.h>
#include <math.h>

typedef __attribute__((ext_vector_type(16))) __bf16 v16bf;
typedef __attribute__((ext_vector_type(8)))  __bf16 bf16x8;
typedef __attribute__((ext_vector_type(4)))  __bf16 bf16x4;
typedef __attribute__((ext_vector_type(8)))  float  v8f;

#define LDT 40           // LDS row stride in bf16 elements (80B, 16B aligned, padded)
#define PROJ_N 1024
#define PROJ_K 1024
#define SCORE_L 256
#define SCORE_K 1024

// Load one 16x16 bf16 WMMA operand fragment from an LDS tile (row-major rows,
// contiguous K). Per ISA 16-bit A layout: lane holds row (lane&15), elements
// 0..7 = K at (lane>>4)*8, elements 8..15 = K at (lane>>4)*8 + 16.
__device__ __forceinline__ v16bf load_frag(const __bf16* row_base, int k0) {
    bf16x8 lo = *(const bf16x8*)(row_base + k0);
    bf16x8 hi = *(const bf16x8*)(row_base + k0 + 16);
    return __builtin_shufflevector(lo, hi, 0,1,2,3,4,5,6,7,8,9,10,11,12,13,14,15);
}

// ---------------------------------------------------------------------------
// Projection: out[m,n] = (sum_k A[m,k]*W[k,n] + bias[n]) * (scale ? scale[n]:1)
// A: (M=16384, K=1024) f32 row-major; W: (K=1024, N=1024) f32 row-major.
// out: bf16 row-major (ld=1024). Software-pipelined: register prefetch of the
// next K-tile overlaps global loads with WMMA; double-buffered LDS gives one
// barrier per K-step.
// ---------------------------------------------------------------------------
__global__ __launch_bounds__(256) void proj_kernel(
    const float* __restrict__ A, const float* __restrict__ W,
    const float* __restrict__ bias, const float* __restrict__ scale,
    __bf16* __restrict__ out)
{
    __shared__ __bf16 As[2][128 * LDT];   // rows = m, cols = k
    __shared__ __bf16 Bs[2][128 * LDT];   // rows = n, cols = k (transposed W tile)

    const int tid  = threadIdx.x;
    const int lane = tid & 31;
    const int wave = tid >> 5;
    const int wm   = wave & 3;          // 4 waves along M  -> 32 rows each
    const int wn   = wave >> 2;         // 2 waves along N  -> 64 cols each
    const int tn   = blockIdx.x & 7;    // N tiles: 1024/128 = 8
    const int tm   = blockIdx.x >> 3;   // M tiles

    // staging geometry (uniform per thread across iterations)
    const int n4 = tid & 31;            // W micro-tile: 4 cols at n4*4
    const int kq = tid >> 5;            //               4 rows at kq*4

    float4 pa[4], pw[4];                // register prefetch of one K-tile

    auto load_tile = [&](int kb) {
        #pragma unroll
        for (int it = 0; it < 4; ++it) {
            int f = tid + it * 256, row = f >> 3, c4 = f & 7;
            pa[it] = *(const float4*)(A + (size_t)(tm * 128 + row) * PROJ_K + kb + c4 * 4);
        }
        #pragma unroll
        for (int e = 0; e < 4; ++e)
            pw[e] = *(const float4*)(W + (size_t)(kb + kq * 4 + e) * PROJ_N + tn * 128 + n4 * 4);
    };
    auto store_tile = [&](int p) {
        #pragma unroll
        for (int it = 0; it < 4; ++it) {
            int f = tid + it * 256, row = f >> 3, c4 = f & 7;
            bf16x4 h = { (__bf16)pa[it].x, (__bf16)pa[it].y,
                         (__bf16)pa[it].z, (__bf16)pa[it].w };
            *(bf16x4*)(&As[p][row * LDT + c4 * 4]) = h;   // single ds_store_b64
        }
        const float* pv = &pw[0].x;                       // 4x4 register transpose
        #pragma unroll
        for (int j = 0; j < 4; ++j) {
            bf16x4 h = { (__bf16)pv[0 * 4 + j], (__bf16)pv[1 * 4 + j],
                         (__bf16)pv[2 * 4 + j], (__bf16)pv[3 * 4 + j] };
            *(bf16x4*)(&Bs[p][(n4 * 4 + j) * LDT + kq * 4]) = h;
        }
    };

    const v8f zero = {0.f,0.f,0.f,0.f,0.f,0.f,0.f,0.f};
    v8f c[2][4];
    #pragma unroll
    for (int mt = 0; mt < 2; ++mt)
        #pragma unroll
        for (int nt = 0; nt < 4; ++nt) c[mt][nt] = zero;

    const int k0 = (lane >> 4) * 8;

    load_tile(0);
    store_tile(0);
    __syncthreads();

    int p = 0;
    for (int kb = 0; kb < PROJ_K; kb += 32) {
        const bool has_next = (kb + 32) < PROJ_K;
        if (has_next) load_tile(kb + 32);      // global loads in flight under WMMA

        v16bf af[2], bfr[4];
        #pragma unroll
        for (int mt = 0; mt < 2; ++mt)
            af[mt] = load_frag(&As[p][(wm * 32 + mt * 16 + (lane & 15)) * LDT], k0);
        #pragma unroll
        for (int nt = 0; nt < 4; ++nt)
            bfr[nt] = load_frag(&Bs[p][(wn * 64 + nt * 16 + (lane & 15)) * LDT], k0);

        #pragma unroll
        for (int mt = 0; mt < 2; ++mt)
            #pragma unroll
            for (int nt = 0; nt < 4; ++nt)
                c[mt][nt] = __builtin_amdgcn_wmma_f32_16x16x32_bf16(
                    false, af[mt], false, bfr[nt], (short)0, c[mt][nt], false, false);

        if (has_next) store_tile(p ^ 1);       // waits loadcnt here, after WMMA issue
        __syncthreads();
        p ^= 1;
    }

    // --- epilogue: bias + optional scale, store bf16 ---
    #pragma unroll
    for (int mt = 0; mt < 2; ++mt) {
        #pragma unroll
        for (int nt = 0; nt < 4; ++nt) {
            int col = tn * 128 + wn * 64 + nt * 16 + (lane & 15);
            float bi = bias[col];
            float sc = scale ? scale[col] : 1.0f;
            #pragma unroll
            for (int r = 0; r < 8; ++r) {
                int row = tm * 128 + wm * 32 + mt * 16 + r + ((lane >> 4) << 3);
                out[(size_t)row * PROJ_N + col] = (__bf16)((c[mt][nt][r] + bi) * sc);
            }
        }
    }
}

// ---------------------------------------------------------------------------
// Scores: per batch, C[i,j] = sum_k paw[i,k]*pb[j,k] + wbias  (f32 out, ld=256)
// paw/pb: bf16 (256 x 1024) per batch. Same pipelined structure; B = pb^T via
// the layout duality (rows of pb staged row-major feed the B operand).
// ---------------------------------------------------------------------------
__global__ __launch_bounds__(256) void score_kernel(
    const __bf16* __restrict__ paw, const __bf16* __restrict__ pb,
    const float* __restrict__ wbias, float* __restrict__ scores)
{
    __shared__ __bf16 As[2][128 * LDT];   // rows = i
    __shared__ __bf16 Bs[2][128 * LDT];   // rows = j

    const int tid  = threadIdx.x;
    const int lane = tid & 31;
    const int wave = tid >> 5;
    const int wm   = wave & 3;
    const int wn   = wave >> 2;
    const int b    = blockIdx.x >> 2;
    const int t    = blockIdx.x & 3;
    const int tileM = (t >> 1) * 128;
    const int tileN = (t & 1) * 128;

    const __bf16* Abase = paw + ((size_t)b * SCORE_L + tileM) * SCORE_K;
    const __bf16* Bbase = pb  + ((size_t)b * SCORE_L + tileN) * SCORE_K;

    bf16x8 ra[2], rb[2];                  // register prefetch (4 x 16B)

    auto load_tile = [&](int kb) {
        #pragma unroll
        for (int it = 0; it < 2; ++it) {
            int f = tid + it * 256, row = f >> 2, c8 = f & 3;
            ra[it] = *(const bf16x8*)(Abase + (size_t)row * SCORE_K + kb + c8 * 8);
            rb[it] = *(const bf16x8*)(Bbase + (size_t)row * SCORE_K + kb + c8 * 8);
        }
    };
    auto store_tile = [&](int p) {
        #pragma unroll
        for (int it = 0; it < 2; ++it) {
            int f = tid + it * 256, row = f >> 2, c8 = f & 3;
            *(bf16x8*)(&As[p][row * LDT + c8 * 8]) = ra[it];
            *(bf16x8*)(&Bs[p][row * LDT + c8 * 8]) = rb[it];
        }
    };

    const v8f zero = {0.f,0.f,0.f,0.f,0.f,0.f,0.f,0.f};
    v8f c[2][4];
    #pragma unroll
    for (int mt = 0; mt < 2; ++mt)
        #pragma unroll
        for (int nt = 0; nt < 4; ++nt) c[mt][nt] = zero;

    const int k0 = (lane >> 4) * 8;

    load_tile(0);
    store_tile(0);
    __syncthreads();

    int p = 0;
    for (int kb = 0; kb < SCORE_K; kb += 32) {
        const bool has_next = (kb + 32) < SCORE_K;
        if (has_next) load_tile(kb + 32);

        v16bf af[2], bfr[4];
        #pragma unroll
        for (int mt = 0; mt < 2; ++mt)
            af[mt] = load_frag(&As[p][(wm * 32 + mt * 16 + (lane & 15)) * LDT], k0);
        #pragma unroll
        for (int nt = 0; nt < 4; ++nt)
            bfr[nt] = load_frag(&Bs[p][(wn * 64 + nt * 16 + (lane & 15)) * LDT], k0);

        #pragma unroll
        for (int mt = 0; mt < 2; ++mt)
            #pragma unroll
            for (int nt = 0; nt < 4; ++nt)
                c[mt][nt] = __builtin_amdgcn_wmma_f32_16x16x32_bf16(
                    false, af[mt], false, bfr[nt], (short)0, c[mt][nt], false, false);

        if (has_next) store_tile(p ^ 1);
        __syncthreads();
        p ^= 1;
    }

    const float wb = wbias[0];
    float* outb = scores + (size_t)b * SCORE_L * SCORE_L;
    #pragma unroll
    for (int mt = 0; mt < 2; ++mt) {
        #pragma unroll
        for (int nt = 0; nt < 4; ++nt) {
            int col = tileN + wn * 64 + nt * 16 + (lane & 15);
            #pragma unroll
            for (int r = 0; r < 8; ++r) {
                int row = tileM + wm * 32 + mt * 16 + r + ((lane >> 4) << 3);
                outb[(size_t)row * SCORE_L + col] = c[mt][nt][r] + wb;
            }
        }
    }
}

// ---------------------------------------------------------------------------
// Softmax over 256*256 = 65536 elements per batch, in place. 1024 threads,
// 64 elements each (coalesced, kept in registers), LDS tree reductions.
// ---------------------------------------------------------------------------
__global__ __launch_bounds__(1024) void softmax_kernel(float* __restrict__ scores)
{
    __shared__ float red[1024];
    const int tid = threadIdx.x;
    float* p = scores + (size_t)blockIdx.x * 65536;

    float v[64];
    float m = -INFINITY;
    #pragma unroll
    for (int i = 0; i < 64; ++i) {
        v[i] = p[tid + i * 1024];
        m = fmaxf(m, v[i]);
    }
    red[tid] = m;
    __syncthreads();
    for (int s = 512; s > 0; s >>= 1) {
        if (tid < s) red[tid] = fmaxf(red[tid], red[tid + s]);
        __syncthreads();
    }
    m = red[0];
    __syncthreads();

    float sum = 0.f;
    #pragma unroll
    for (int i = 0; i < 64; ++i) {
        v[i] = __expf(v[i] - m);
        sum += v[i];
    }
    red[tid] = sum;
    __syncthreads();
    for (int s = 512; s > 0; s >>= 1) {
        if (tid < s) red[tid] += red[tid + s];
        __syncthreads();
    }
    const float inv = 1.0f / red[0];
    #pragma unroll
    for (int i = 0; i < 64; ++i)
        p[tid + i * 1024] = v[i] * inv;
}

extern "C" void kernel_launch(void* const* d_in, const int* in_sizes, int n_in,
                              void* d_out, int out_size, void* d_ws, size_t ws_size,
                              hipStream_t stream) {
    (void)in_sizes; (void)n_in; (void)out_size; (void)ws_size;
    const float* a     = (const float*)d_in[0];
    const float* b     = (const float*)d_in[1];
    const float* Wa    = (const float*)d_in[2];
    const float* ba    = (const float*)d_in[3];
    const float* Wb    = (const float*)d_in[4];
    const float* bb    = (const float*)d_in[5];
    const float* w     = (const float*)d_in[6];
    const float* wbias = (const float*)d_in[7];

    const size_t proj_elems = (size_t)64 * 256 * 1024;         // 16.7M bf16 each
    __bf16* paw = (__bf16*)d_ws;                               // (pa + ba) * w, bf16
    __bf16* pbp = (__bf16*)((char*)d_ws + proj_elems * 2);     // pb + bb, bf16

    // M/128 * N/128 = 128 * 8 = 1024 workgroups, 256 threads (8 wave32s)
    proj_kernel<<<dim3(1024), dim3(256), 0, stream>>>(a, Wa, ba, w,       paw);
    proj_kernel<<<dim3(1024), dim3(256), 0, stream>>>(b, Wb, bb, nullptr, pbp);
    // 64 batches * 4 tiles of 128x128
    score_kernel<<<dim3(256), dim3(256), 0, stream>>>(paw, pbp, wbias, (float*)d_out);
    softmax_kernel<<<dim3(64), dim3(1024), 0, stream>>>((float*)d_out);
}
